// FinalCoarseToFineDensenSampleIGAModulev3_2_9182640078990
// MI455X (gfx1250) — matrix operand order repaired and checked by reading, not compiled, not run on hardware
//
#include <hip/hip_runtime.h>
#include <hip/hip_bf16.h>
#include <math.h>

// ---------------------------------------------------------------------------
// Problem constants (match reference)
// ---------------------------------------------------------------------------
constexpr int B_   = 2;
constexpr int N_   = 8192;
constexpr int K_   = 1024;
constexpr int C_   = 512;
constexpr int C2_  = 1024;            // 2*C
constexpr int MTOT = B_ * N_;         // 16384 query rows
constexpr int RTOT = B_ * K_;         // 2048 parent rows
constexpr int OUTC = 3 + 9 + C_;      // 524 output channels per node
constexpr int KCH  = 256;             // GEMM K-chunk staged in LDS (64x256 f16 = 32KB)

typedef __attribute__((ext_vector_type(16))) _Float16 v16h;
typedef __attribute__((ext_vector_type(8)))  _Float16 v8h;
typedef __attribute__((ext_vector_type(8)))  float    v8f;
typedef __attribute__((ext_vector_type(4)))  unsigned int v4u;
typedef __attribute__((ext_vector_type(8)))  int          v8i;
typedef __attribute__((ext_vector_type(4)))  int          v4i;

union V16H { v16h v; v8h h[2]; };

#if __has_builtin(__builtin_amdgcn_tensor_load_to_lds)
#define HAVE_TDM 1
#else
#define HAVE_TDM 0
#endif

// ---------------------------------------------------------------------------
// WMMA fragment loaders (contiguous K order, same for A and B, wave32)
// A: 16x32 f16, row-major source [*, lda]. lane&15 = row M; lanes<16 hold
//    K 0..15, lanes>=16 hold K 16..31.
// B: 32x16 f16 from a row-major "B-transposed" matrix Bt[Ncols, Kd]:
//    lane&15 = output col; lane>>4 selects K half (16 contiguous K values).
// ---------------------------------------------------------------------------
__device__ __forceinline__ v16h load_a_frag(const _Float16* __restrict__ A,
                                            int lda, int row0, int k0, int lane) {
  const int r  = row0 + (lane & 15);
  const int kb = k0 + ((lane >> 4) << 4);
  const _Float16* p = A + (size_t)r * lda + kb;
  V16H u;
  u.h[0] = *(const v8h*)(p);
  u.h[1] = *(const v8h*)(p + 8);
  return u.v;
}

__device__ __forceinline__ v16h load_b_frag(const _Float16* __restrict__ Bt,
                                            int ldb, int col0, int k0, int lane) {
  const int c  = col0 + (lane & 15);
  const int kb = k0 + ((lane >> 4) << 4);
  return *(const v16h*)(Bt + (size_t)c * ldb + kb);
}

__device__ __forceinline__ float gelu_exact(float x) {
  return 0.5f * x * (1.0f + erff(x * 0.70710678118654752f));
}

// ---------------------------------------------------------------------------
// Tensor Data Mover: stage a 2D [tile_d1 x tile_d0] f16 slab (row stride
// stride0 elements) from global into LDS.  D# built per ISA 08_async_tensor §8:
//   group0: count=1 | lds_addr | global_addr[56:0] | type=2
//   group1: data_size=2B, tensor/tile dims, tensor_dim0_stride
//   groups 2/3 zero (<=2D tensor)
// ---------------------------------------------------------------------------
__device__ __forceinline__ void tdm_stage_2d_f16(
    unsigned lds_byte_addr, const _Float16* gsrc,
    unsigned tile_d0, unsigned tile_d1, unsigned long long stride0_elems) {
#if HAVE_TDM
  const unsigned long long ga = (unsigned long long)(uintptr_t)gsrc;
  v4u g0;
  g0[0] = 1u;                                              // count=1, user mode
  g0[1] = lds_byte_addr;                                   // lds_addr (bytes)
  g0[2] = (unsigned)(ga & 0xffffffffu);                    // global_addr lo
  g0[3] = (unsigned)((ga >> 32) & 0x01ffffffu) | (2u << 30); // addr hi + type=2
  v8i g1;
  g1[0] = (int)(1u << 16);                                 // mask=0, data_size=1 (2B)
  g1[1] = (int)((tile_d0 & 0xffffu) << 16);                // tensor_dim0 lo16
  g1[2] = (int)((tile_d0 >> 16) | ((tile_d1 & 0xffffu) << 16)); // dim0 hi | dim1 lo
  g1[3] = (int)((tile_d1 >> 16) | ((tile_d0 & 0xffffu) << 16)); // dim1 hi | tile_dim0
  g1[4] = (int)(tile_d1 & 0xffffu);                        // tile_dim1, tile_dim2=0
  g1[5] = (int)(stride0_elems & 0xffffffffull);            // stride0 lo
  g1[6] = (int)((stride0_elems >> 32) & 0xffffull);        // stride0 hi | stride1=0
  g1[7] = 0;
  v4i z4 = {0, 0, 0, 0};
#if defined(__clang_major__) && (__clang_major__ >= 23)
  v8i z8 = {0, 0, 0, 0, 0, 0, 0, 0};
  __builtin_amdgcn_tensor_load_to_lds(g0, g1, z4, z4, z8, 0);
#else
  __builtin_amdgcn_tensor_load_to_lds(g0, g1, z4, z4, 0);
#endif
  __builtin_amdgcn_s_wait_tensorcnt(0);
#else
  (void)lds_byte_addr; (void)gsrc; (void)tile_d0; (void)tile_d1; (void)stride0_elems;
#endif
}

// ---------------------------------------------------------------------------
// Weight prep: fold LN affine into weights.  Wout = f16(W * g[col]),
// biasOut[row] = sum_j W[row,j]*bvec[j] + (bias0 ? bias0[row] : 0)
// ---------------------------------------------------------------------------
__global__ __launch_bounds__(256) void prep_weights_kernel(
    const float* __restrict__ W, const float* __restrict__ g,
    const float* __restrict__ bvec, const float* __restrict__ bias0,
    _Float16* __restrict__ Wout, float* __restrict__ biasOut, int Kd) {
  __shared__ float rb[8];
  const int row = blockIdx.x;
  const float* wr = W + (size_t)row * Kd;
  float p = 0.0f;
  for (int j = threadIdx.x; j < Kd; j += 256) {
    float w = wr[j];
    Wout[(size_t)row * Kd + j] = (_Float16)(w * g[j]);
    p += w * bvec[j];
  }
  for (int o = 16; o > 0; o >>= 1) p += __shfl_down(p, o, 32);
  const int lane = threadIdx.x & 31, wave = threadIdx.x >> 5;
  if (lane == 0) rb[wave] = p;
  __syncthreads();
  if (threadIdx.x == 0) {
    float s = 0.0f;
    for (int w = 0; w < 8; ++w) s += rb[w];
    biasOut[row] = s + (bias0 ? bias0[row] : 0.0f);
  }
}

// ---------------------------------------------------------------------------
// LayerNorm x-hat (gamma/beta folded elsewhere). One block per row, C=512.
// ---------------------------------------------------------------------------
__global__ __launch_bounds__(256) void ln_xhat_kernel(
    const float* __restrict__ X, _Float16* __restrict__ Xh) {
  __shared__ float xs[C_];
  __shared__ float rb[16];
  const int row = blockIdx.x;
  const float* xr = X + (size_t)row * C_;
  float s = 0.0f, ss = 0.0f;
  for (int j = threadIdx.x; j < C_; j += 256) {
    float v = xr[j]; xs[j] = v; s += v; ss += v * v;
  }
  for (int o = 16; o > 0; o >>= 1) { s += __shfl_down(s, o, 32); ss += __shfl_down(ss, o, 32); }
  const int lane = threadIdx.x & 31, wave = threadIdx.x >> 5;
  if (lane == 0) { rb[wave] = s; rb[8 + wave] = ss; }
  __syncthreads();
  if (threadIdx.x == 0) {
    float S = 0.0f, SS = 0.0f;
    for (int w = 0; w < 8; ++w) { S += rb[w]; SS += rb[8 + w]; }
    float mean = S / (float)C_;
    float var  = SS / (float)C_ - mean * mean;
    rb[0] = mean; rb[1] = rsqrtf(var + 1e-5f);
  }
  __syncthreads();
  const float mean = rb[0], rstd = rb[1];
  for (int j = threadIdx.x; j < C_; j += 256)
    Xh[(size_t)row * C_ + j] = (_Float16)((xs[j] - mean) * rstd);
}

// ---------------------------------------------------------------------------
// Generic WMMA GEMM: D[M,Nc] = A[M,Kd] @ Bt[Nc,Kd]^T + bias
// Block = 128 (4 waves).  Block tile = 64 rows x 64 cols; each wave owns
// 16 rows x 64 cols (4 accumulators -> each A frag feeds 4 WMMAs).
// The 64-col B slab is staged K-chunk by K-chunk into LDS via the TDM
// (tensor_load_to_lds + s_wait_tensorcnt), shared by all 4 waves.
// EPI 0: store f16 | 1: +mu_p@Wpos^T, store f16 | 2: store f32 | 3: gelu, f16
// grid = (M/64, Nc/64); dynamic LDS = 64*KCH*2 bytes.
// ---------------------------------------------------------------------------
template<int EPI>
__global__ __launch_bounds__(128) void wmma_gemm_kernel(
    const _Float16* __restrict__ A, const _Float16* __restrict__ Bt,
    const float* __restrict__ bias,
    _Float16* __restrict__ outH, float* __restrict__ outF,
    const float* __restrict__ mu3, const float* __restrict__ Wpos,
    int Nc, int Kd) {
  extern __shared__ _Float16 sB[];                 // [64][KCH]
  const int lane  = threadIdx.x & 31;
  const int wave  = threadIdx.x >> 5;
  const int row0  = blockIdx.x * 64 + wave * 16;
  const int col0g = blockIdx.y * 64;

  v8f acc[4];
#pragma unroll
  for (int t = 0; t < 4; ++t) {
    const float bv = bias ? bias[col0g + t * 16 + (lane & 15)] : 0.0f;
#pragma unroll
    for (int g = 0; g < 8; ++g) acc[t][g] = bv;
  }

  for (int kc = 0; kc < Kd; kc += KCH) {
    __syncthreads();                                // previous chunk consumed
#if HAVE_TDM
    if (threadIdx.x == 0) {
      tdm_stage_2d_f16((unsigned)(uintptr_t)(&sB[0]),
                       Bt + (size_t)col0g * Kd + kc,
                       (unsigned)KCH, 64u, (unsigned long long)Kd);
    }
#else
    for (int i = threadIdx.x * 8; i < 64 * KCH; i += 128 * 8) {
      const int r = i / KCH, cc = i % KCH;
      *(v8h*)&sB[i] = *(const v8h*)(Bt + (size_t)(col0g + r) * Kd + kc + cc);
    }
#endif
    __syncthreads();                                // slab visible to all waves

#pragma unroll 2
    for (int k0 = 0; k0 < KCH; k0 += 32) {
      if (kc + k0 + 32 < Kd)
        __builtin_prefetch(A + (size_t)(row0 + (lane & 15)) * Kd + kc + k0 + 32, 0, 1);
      v16h af = load_a_frag(A, Kd, row0, kc + k0, lane);
#pragma unroll
      for (int t = 0; t < 4; ++t) {
        v16h bf = load_b_frag(sB, KCH, t * 16, k0, lane);   // ds_load path
        acc[t] = __builtin_amdgcn_wmma_f32_16x16x32_f16(
            false, af, false, bf, (short)0, acc[t], false, false);
      }
    }
  }

  const int mbase = row0 + ((lane >> 4) << 3);
#pragma unroll
  for (int t = 0; t < 4; ++t) {
    const int ncol = col0g + t * 16 + (lane & 15);
#pragma unroll
    for (int g = 0; g < 8; ++g) {
      const int m = mbase + g;
      float val = acc[t][g];
      if constexpr (EPI == 1) {
        val += mu3[m * 3 + 0] * Wpos[ncol * 3 + 0]
             + mu3[m * 3 + 1] * Wpos[ncol * 3 + 1]
             + mu3[m * 3 + 2] * Wpos[ncol * 3 + 2];
      }
      if constexpr (EPI == 3) val = gelu_exact(val);
      if constexpr (EPI == 2) outF[(size_t)m * Nc + ncol] = val;
      else                    outH[(size_t)m * Nc + ncol] = (_Float16)val;
    }
  }
}

// ---------------------------------------------------------------------------
// Logits + mask + argmax + softmax usage. One block = 16 query rows.
// 8 waves x 8 col-tiles = full K=1024; logits tile kept in 64KB dynamic LDS.
// ---------------------------------------------------------------------------
__global__ __launch_bounds__(256) void logits_kernel(
    const _Float16* __restrict__ q, const _Float16* __restrict__ kmat,
    const float* __restrict__ mask_parent, const float* __restrict__ node_mask,
    int* __restrict__ idx_out, float* __restrict__ usage_acc) {
  extern __shared__ float sl[];                 // [16][K_]
  const int lane = threadIdx.x & 31;
  const int wave = threadIdx.x >> 5;
  const int row0 = blockIdx.x * 16;             // global query row base
  const int b    = row0 / N_;
  const _Float16* kb = kmat + (size_t)b * K_ * C_;
  const float invs = 1.0f / (sqrtf((float)C_) + 1e-6f);

  for (int t = 0; t < 8; ++t) {
    const int col0 = (wave * 8 + t) * 16;
    v8f acc;
#pragma unroll
    for (int g = 0; g < 8; ++g) acc[g] = 0.0f;
    for (int k0 = 0; k0 < C_; k0 += 32) {
      v16h af = load_a_frag(q,  C_, row0, k0, lane);
      v16h bf = load_b_frag(kb, C_, col0, k0, lane);
      acc = __builtin_amdgcn_wmma_f32_16x16x32_f16(
          false, af, false, bf, (short)0, acc, false, false);
    }
    const int mbase = (lane >> 4) << 3;
    const int n     = col0 + (lane & 15);
#pragma unroll
    for (int g = 0; g < 8; ++g)
      sl[(mbase + g) * K_ + n] = acc[g] * invs;
  }
  __syncthreads();

  // per-row: mask, argmax, softmax (2 rows per wave)
  for (int rr = 0; rr < 2; ++rr) {
    const int r    = wave * 2 + rr;
    const int grow = row0 + r;
    const float nm = node_mask[grow];
    float best = -3.4e38f; int bidx = 0;
    for (int c = lane; c < K_; c += 32) {
      float val = sl[r * K_ + c];
      if (mask_parent[b * K_ + c] < 0.5f) val = -10000.0f;
      if (nm < 0.5f)                      val = -10000.0f;
      sl[r * K_ + c] = val;
      if (val > best) { best = val; bidx = c; }
    }
    for (int o = 16; o > 0; o >>= 1) {
      float ob = __shfl_down(best, o, 32);
      int   oi = __shfl_down(bidx, o, 32);
      if (ob > best || (ob == best && oi < bidx)) { best = ob; bidx = oi; }
    }
    best = __shfl(best, 0, 32);
    bidx = __shfl(bidx, 0, 32);
    if (lane == 0) idx_out[grow] = bidx;

    float se = 0.0f;
    for (int c = lane; c < K_; c += 32) se += expf(sl[r * K_ + c] - best);
    for (int o = 16; o > 0; o >>= 1) se += __shfl_down(se, o, 32);
    se = __shfl(se, 0, 32);
    const float inv = 1.0f / se;
    for (int c = lane; c < K_; c += 32)
      sl[r * K_ + c] = expf(sl[r * K_ + c] - best) * inv;
  }
  __syncthreads();

  // column-wise partial usage: one atomic per column per block
  for (int c = threadIdx.x; c < K_; c += 256) {
    float s = 0.0f;
    for (int m = 0; m < 16; ++m) s += sl[m * K_ + c];
    atomicAdd(&usage_acc[b * K_ + c], s);
  }
}

// ---------------------------------------------------------------------------
// Gather (hard routing) + feat LN x-hat + s_base/mu/Sig extraction.
// One block per node row; feat = [s_trunk, v[idx]*nm] (2C = 1024).
// ---------------------------------------------------------------------------
__global__ __launch_bounds__(256) void gather_feat_kernel(
    const float* __restrict__ st, const float* __restrict__ vmat,
    const float* __restrict__ mu_p, const float* __restrict__ Sig_p,
    const float* __restrict__ node_mask, const int* __restrict__ idx,
    _Float16* __restrict__ xhat_feat, float* __restrict__ out,
    float* __restrict__ mu_base, float* __restrict__ scale_base,
    float* __restrict__ sig_base) {
  __shared__ float fb[C2_];
  __shared__ float rb[16];
  const int row = blockIdx.x;
  const int b   = row / N_;
  const int id  = idx[row];
  const float nm = node_mask[row];
  const float* str = st + (size_t)row * C_;
  const float* vr  = vmat + ((size_t)b * K_ + id) * C_;

  float s = 0.0f, ss = 0.0f;
  for (int j = threadIdx.x; j < C2_; j += 256) {
    float x = (j < C_) ? str[j] : vr[j - C_] * nm;
    fb[j] = x; s += x; ss += x * x;
    if (j >= C_) out[(size_t)row * OUTC + 12 + (j - C_)] = x;  // s_base
  }
  for (int o = 16; o > 0; o >>= 1) { s += __shfl_down(s, o, 32); ss += __shfl_down(ss, o, 32); }
  const int lane = threadIdx.x & 31, wave = threadIdx.x >> 5;
  if (lane == 0) { rb[wave] = s; rb[8 + wave] = ss; }
  __syncthreads();
  if (threadIdx.x == 0) {
    float S = 0.0f, SS = 0.0f;
    for (int w = 0; w < 8; ++w) { S += rb[w]; SS += rb[8 + w]; }
    float mean = S / (float)C2_;
    float var  = SS / (float)C2_ - mean * mean;
    rb[0] = mean; rb[1] = rsqrtf(var + 1e-5f);
  }
  __syncthreads();
  const float mean = rb[0], rstd = rb[1];
  for (int j = threadIdx.x; j < C2_; j += 256)
    xhat_feat[(size_t)row * C2_ + j] = (_Float16)((fb[j] - mean) * rstd);

  if (threadIdx.x < 3)
    mu_base[row * 3 + threadIdx.x] = mu_p[((size_t)b * K_ + id) * 3 + threadIdx.x] * nm;
  if (threadIdx.x < 9) {
    const int i = threadIdx.x / 3, j2 = threadIdx.x % 3;
    const float* Sp = Sig_p + ((size_t)b * K_ + id) * 9;
    float sv = 0.5f * (Sp[i * 3 + j2] + Sp[j2 * 3 + i]) * nm;   // _sym
    sig_base[(size_t)row * 9 + threadIdx.x] = sv;
    if (i == j2) scale_base[row * 3 + i] = sqrtf(fmaxf(sv, 1e-8f));
  }
}

// ---------------------------------------------------------------------------
// Head: raw_offset / log_s_delta (3-wide projections), mu0 & Sig0 epilogue.
// One wave per row; 8 rows per block.
// ---------------------------------------------------------------------------
__global__ __launch_bounds__(256) void head_kernel(
    const _Float16* __restrict__ hd, const _Float16* __restrict__ hs,
    const float* __restrict__ Wd2, const float* __restrict__ bd2,
    const float* __restrict__ Ws2, const float* __restrict__ bs2,
    const float* __restrict__ mu_base, const float* __restrict__ scale_base,
    const float* __restrict__ sig_base, const float* __restrict__ node_mask,
    float* __restrict__ out, float* __restrict__ center_acc) {
  const int wave = threadIdx.x >> 5, lane = threadIdx.x & 31;
  const int row  = blockIdx.x * 8 + wave;
  const _Float16* hr  = hd + (size_t)row * C_;
  const _Float16* hsr = hs + (size_t)row * C_;
  float pd0 = 0, pd1 = 0, pd2 = 0, ps0 = 0, ps1 = 0, ps2 = 0;
  for (int c = lane; c < C_; c += 32) {
    float a  = (float)hr[c];
    float bb = (float)hsr[c];
    pd0 += a * Wd2[c]; pd1 += a * Wd2[C_ + c]; pd2 += a * Wd2[2 * C_ + c];
    ps0 += bb * Ws2[c]; ps1 += bb * Ws2[C_ + c]; ps2 += bb * Ws2[2 * C_ + c];
  }
  for (int o = 16; o > 0; o >>= 1) {
    pd0 += __shfl_down(pd0, o, 32); pd1 += __shfl_down(pd1, o, 32);
    pd2 += __shfl_down(pd2, o, 32); ps0 += __shfl_down(ps0, o, 32);
    ps1 += __shfl_down(ps1, o, 32); ps2 += __shfl_down(ps2, o, 32);
  }
  if (lane == 0) {
    const int b = row / N_;
    const float nm = node_mask[row];
    float d3[3] = { tanhf(pd0 + bd2[0]), tanhf(pd1 + bd2[1]), tanhf(pd2 + bd2[2]) };
    float sd[3] = { expf(0.5f * tanhf(ps0 + bs2[0])),
                    expf(0.5f * tanhf(ps1 + bs2[1])),
                    expf(0.5f * tanhf(ps2 + bs2[2])) };
    float* orow = out + (size_t)row * OUTC;
    for (int d = 0; d < 3; ++d) {
      float m0 = (mu_base[row * 3 + d] + d3[d] * scale_base[row * 3 + d]) * nm;
      orow[d] = m0;
      atomicAdd(&center_acc[b * 3 + d], m0);
    }
    for (int i = 0; i < 3; ++i)
      for (int j = 0; j < 3; ++j) {
        float sv = (sig_base[(size_t)row * 9 + i * 3 + j] * sd[i] * sd[j]
                    + ((i == j) ? 1e-6f : 0.0f)) * nm;
        orow[3 + i * 3 + j] = sv;
      }
  }
}

// ---------------------------------------------------------------------------
// Pass 2 over mu0: accumulate per-batch dist sums and node_mask sum.
// ---------------------------------------------------------------------------
__global__ __launch_bounds__(256) void dist_kernel(
    const float* __restrict__ out, const float* __restrict__ node_mask,
    const float* __restrict__ center_acc, float* __restrict__ scale_acc) {
  const int row = blockIdx.x * blockDim.x + threadIdx.x;
  if (row >= MTOT) return;
  const int b = row / N_;
  const float invN = 1.0f / (float)N_;
  float dx = out[(size_t)row * OUTC + 0] - center_acc[b * 3 + 0] * invN;
  float dy = out[(size_t)row * OUTC + 1] - center_acc[b * 3 + 1] * invN;
  float dz = out[(size_t)row * OUTC + 2] - center_acc[b * 3 + 2] * invN;
  const float dist = sqrtf(dx * dx + dy * dy + dz * dz);
  const float nm = node_mask[row];
  atomicAdd(&scale_acc[b], dist * nm);
  atomicAdd(&scale_acc[2], nm);
}

// ---------------------------------------------------------------------------
// Final scalar regularizer.
// ---------------------------------------------------------------------------
__global__ __launch_bounds__(256) void reg_kernel(
    const float* __restrict__ usage_acc, const float* __restrict__ scale_acc,
    float* __restrict__ reg) {
  __shared__ float rb[256];
  float p = 0.0f;
  for (int i = threadIdx.x; i < B_ * K_; i += 256) {
    float u = usage_acc[i] / (float)N_;
    float d = u - 1.0f / (float)K_;
    p += d * d;
  }
  rb[threadIdx.x] = p;
  __syncthreads();
  for (int s = 128; s > 0; s >>= 1) {
    if (threadIdx.x < s) rb[threadIdx.x] += rb[threadIdx.x + s];
    __syncthreads();
  }
  if (threadIdx.x == 0) {
    const float loss_bal = rb[0] / (float)(B_ * K_);
    const float msum = fmaxf(scale_acc[2], 1.0f);
    float ls = 0.0f;
    for (int b = 0; b < B_; ++b) {
      float cs = scale_acc[b] / msum;
      ls += fmaxf(0.5f - cs, 0.0f);
    }
    ls /= (float)B_;
    reg[0] = 2.0f * loss_bal + ls;
  }
}

__global__ __launch_bounds__(256) void add_reg_kernel(
    float* __restrict__ out, const float* __restrict__ reg, int n) {
  const int i = blockIdx.x * blockDim.x + threadIdx.x;
  if (i < n) out[i] += reg[0];
}

// ---------------------------------------------------------------------------
// Host launch
// ---------------------------------------------------------------------------
extern "C" void kernel_launch(void* const* d_in, const int* in_sizes, int n_in,
                              void* d_out, int out_size, void* d_ws, size_t ws_size,
                              hipStream_t stream) {
  (void)in_sizes; (void)n_in; (void)ws_size;
  const float* s_trunk     = (const float*)d_in[0];
  const float* s_parent    = (const float*)d_in[1];
  const float* mu_p        = (const float*)d_in[2];
  const float* Sig_p       = (const float*)d_in[3];
  const float* mask_parent = (const float*)d_in[4];
  const float* node_mask   = (const float*)d_in[5];
  const float* ln_q_g = (const float*)d_in[6],  *ln_q_b = (const float*)d_in[7];
  const float* ln_k_g = (const float*)d_in[8],  *ln_k_b = (const float*)d_in[9];
  const float* ln_v_g = (const float*)d_in[10], *ln_v_b = (const float*)d_in[11];
  const float* Wq   = (const float*)d_in[12];
  const float* Wk   = (const float*)d_in[13];
  const float* Wv   = (const float*)d_in[14];
  const float* Wpos = (const float*)d_in[15];
  const float* d_ln_g = (const float*)d_in[16], *d_ln_b = (const float*)d_in[17];
  const float* Wd1 = (const float*)d_in[18], *bd1 = (const float*)d_in[19];
  const float* Wd2 = (const float*)d_in[20], *bd2 = (const float*)d_in[21];
  const float* s_ln_g = (const float*)d_in[22], *s_ln_b = (const float*)d_in[23];
  const float* Ws1 = (const float*)d_in[24], *bs1 = (const float*)d_in[25];
  const float* Ws2 = (const float*)d_in[26], *bs2 = (const float*)d_in[27];
  float* out = (float*)d_out;

  // ---- workspace carve-up -------------------------------------------------
  char* wp = (char*)d_ws;
  auto alloc = [&](size_t bytes) -> void* {
    void* p = (void*)wp;
    wp += (bytes + 255) & ~(size_t)255;
    return p;
  };
  _Float16* xhat_st   = (_Float16*)alloc((size_t)MTOT * C_  * sizeof(_Float16));
  _Float16* xhat_sp   = (_Float16*)alloc((size_t)RTOT * C_  * sizeof(_Float16));
  _Float16* qH        = (_Float16*)alloc((size_t)MTOT * C_  * sizeof(_Float16));
  _Float16* kH        = (_Float16*)alloc((size_t)RTOT * C_  * sizeof(_Float16));
  float*    vF        = (float*)   alloc((size_t)RTOT * C_  * sizeof(float));
  _Float16* WqH       = (_Float16*)alloc((size_t)C_ * C_  * sizeof(_Float16));
  _Float16* WkH       = (_Float16*)alloc((size_t)C_ * C_  * sizeof(_Float16));
  _Float16* WvH       = (_Float16*)alloc((size_t)C_ * C_  * sizeof(_Float16));
  _Float16* Wd1H      = (_Float16*)alloc((size_t)C_ * C2_ * sizeof(_Float16));
  _Float16* Ws1H      = (_Float16*)alloc((size_t)C_ * C2_ * sizeof(_Float16));
  float* biasq  = (float*)alloc(C_ * sizeof(float));
  float* biask  = (float*)alloc(C_ * sizeof(float));
  float* biasv  = (float*)alloc(C_ * sizeof(float));
  float* biasd1 = (float*)alloc(C_ * sizeof(float));
  float* biass1 = (float*)alloc(C_ * sizeof(float));
  int*   idx    = (int*)  alloc((size_t)MTOT * sizeof(int));
  _Float16* xhat_feat = (_Float16*)alloc((size_t)MTOT * C2_ * sizeof(_Float16));
  _Float16* hdH       = (_Float16*)alloc((size_t)MTOT * C_  * sizeof(_Float16));
  _Float16* hsH       = (_Float16*)alloc((size_t)MTOT * C_  * sizeof(_Float16));
  float* mu_base    = (float*)alloc((size_t)MTOT * 3 * sizeof(float));
  float* scale_base = (float*)alloc((size_t)MTOT * 3 * sizeof(float));
  float* sig_base   = (float*)alloc((size_t)MTOT * 9 * sizeof(float));
  // accumulators (zeroed each call): usage[R] | center[8] | scale[4] | reg[4]
  const size_t accum_floats = (size_t)RTOT + 8 + 4 + 4;
  float* accum      = (float*)alloc(accum_floats * sizeof(float));
  float* usage_acc  = accum;
  float* center_acc = accum + RTOT;
  float* scale_acc  = accum + RTOT + 8;
  float* reg        = accum + RTOT + 8 + 4;

  const size_t gemm_lds = (size_t)64 * KCH * sizeof(_Float16);   // 32 KB

  // ---- 1. fold LN affine into weights ------------------------------------
  prep_weights_kernel<<<C_, 256, 0, stream>>>(Wq,  ln_q_g, ln_q_b, nullptr, WqH,  biasq,  C_);
  prep_weights_kernel<<<C_, 256, 0, stream>>>(Wk,  ln_k_g, ln_k_b, nullptr, WkH,  biask,  C_);
  prep_weights_kernel<<<C_, 256, 0, stream>>>(Wv,  ln_v_g, ln_v_b, nullptr, WvH,  biasv,  C_);
  prep_weights_kernel<<<C_, 256, 0, stream>>>(Wd1, d_ln_g, d_ln_b, bd1,     Wd1H, biasd1, C2_);
  prep_weights_kernel<<<C_, 256, 0, stream>>>(Ws1, s_ln_g, s_ln_b, bs1,     Ws1H, biass1, C2_);

  // ---- 2. LN x-hats -------------------------------------------------------
  ln_xhat_kernel<<<MTOT, 256, 0, stream>>>(s_trunk,  xhat_st);
  ln_xhat_kernel<<<RTOT, 256, 0, stream>>>(s_parent, xhat_sp);

  // ---- 3. q/k/v projections (WMMA, TDM-staged B) --------------------------
  wmma_gemm_kernel<0><<<dim3(MTOT / 64, C_ / 64), 128, gemm_lds, stream>>>(
      xhat_st, WqH, biasq, qH, nullptr, nullptr, nullptr, C_, C_);
  wmma_gemm_kernel<1><<<dim3(RTOT / 64, C_ / 64), 128, gemm_lds, stream>>>(
      xhat_sp, WkH, biask, kH, nullptr, mu_p, Wpos, C_, C_);
  wmma_gemm_kernel<2><<<dim3(RTOT / 64, C_ / 64), 128, gemm_lds, stream>>>(
      xhat_sp, WvH, biasv, nullptr, vF, nullptr, nullptr, C_, C_);

  // ---- 4. logits + argmax + usage -----------------------------------------
  hipMemsetAsync(accum, 0, accum_floats * sizeof(float), stream);
  logits_kernel<<<MTOT / 16, 256, (size_t)16 * K_ * sizeof(float), stream>>>(
      qH, kH, mask_parent, node_mask, idx, usage_acc);

  // ---- 5. gather + feature LN ---------------------------------------------
  gather_feat_kernel<<<MTOT, 256, 0, stream>>>(
      s_trunk, vF, mu_p, Sig_p, node_mask, idx,
      xhat_feat, out, mu_base, scale_base, sig_base);

  // ---- 6. two MLP-1 GEMMs (WMMA + gelu, TDM-staged B) ---------------------
  wmma_gemm_kernel<3><<<dim3(MTOT / 64, C_ / 64), 128, gemm_lds, stream>>>(
      xhat_feat, Wd1H, biasd1, hdH, nullptr, nullptr, nullptr, C_, C2_);
  wmma_gemm_kernel<3><<<dim3(MTOT / 64, C_ / 64), 128, gemm_lds, stream>>>(
      xhat_feat, Ws1H, biass1, hsH, nullptr, nullptr, nullptr, C_, C2_);

  // ---- 7. heads + regularizer + broadcast-add ------------------------------
  head_kernel<<<MTOT / 8, 256, 0, stream>>>(
      hdH, hsH, Wd2, bd2, Ws2, bs2,
      mu_base, scale_base, sig_base, node_mask, out, center_acc);
  dist_kernel<<<(MTOT + 255) / 256, 256, 0, stream>>>(out, node_mask, center_acc, scale_acc);
  reg_kernel<<<1, 256, 0, stream>>>(usage_acc, scale_acc, reg);
  add_reg_kernel<<<(out_size + 255) / 256, 256, 0, stream>>>(out, reg, out_size);
}
